// _AttentionBlockModel_17532056502554
// MI455X (gfx1250) — matrix operand order repaired and checked by reading, not compiled, not run on hardware
//
#include <hip/hip_runtime.h>
#include <hip/hip_bf16.h>
#include <hip/hip_fp16.h>

// ---- Shapes (fixed by the reference) ----
#define NB    8
#define SEQ   1024
#define HID   2048
#define NQH   16
#define NKVH  4
#define DH    128
#define NTOK  (NB*SEQ)        // 8192

typedef __attribute__((ext_vector_type(16))) _Float16 v16h;
typedef __attribute__((ext_vector_type(8)))  float    v8f;
typedef __attribute__((ext_vector_type(4)))  int      v4i;
typedef __attribute__((address_space(1)))    v4i      gv4i;   // global int4
typedef __attribute__((address_space(3)))    v4i      lv4i;   // LDS int4

union Frag16 { v16h v; uint4 q[2]; };   // 32 bytes = one f16 WMMA A or B operand

// ---- CDNA5 feature probes -------------------------------------------------
#if defined(__has_builtin)
#  if __has_builtin(__builtin_amdgcn_global_load_async_to_lds_b128) && \
      __has_builtin(__builtin_amdgcn_s_wait_asynccnt)
#    define ASYNC_LDS 1
#  endif
#  if __has_builtin(__builtin_amdgcn_sched_barrier)
#    define SCHED_BARRIER() __builtin_amdgcn_sched_barrier(0)
#  endif
#endif
#ifndef ASYNC_LDS
#  define ASYNC_LDS 0
#endif
#ifndef SCHED_BARRIER
#  define SCHED_BARRIER()
#endif

// -------------------------------------------------------------------------
// f32 -> f16 elementwise convert
// -------------------------------------------------------------------------
__global__ void cvt_f16k(const float* __restrict__ in, _Float16* __restrict__ out, int n) {
    for (int i = blockIdx.x * blockDim.x + threadIdx.x; i < n; i += gridDim.x * blockDim.x)
        out[i] = (_Float16)in[i];
}

// -------------------------------------------------------------------------
// f32 [K][N] -> f16 [N][K] transpose-convert (weights become B-fragment friendly)
// block (32,8), grid (N/32, K/32)
// -------------------------------------------------------------------------
__global__ void transpose_cvt(const float* __restrict__ in, _Float16* __restrict__ out,
                              int K, int N) {
    __shared__ float tile[32][33];
    const int n0 = blockIdx.x * 32, k0 = blockIdx.y * 32;
    const int tx = threadIdx.x, ty = threadIdx.y;
#pragma unroll
    for (int j = 0; j < 32; j += 8)
        tile[ty + j][tx] = in[(size_t)(k0 + ty + j) * N + n0 + tx];
    __syncthreads();
#pragma unroll
    for (int j = 0; j < 32; j += 8)
        out[(size_t)(n0 + ty + j) * K + k0 + tx] = (_Float16)tile[tx][ty + j];
}

// -------------------------------------------------------------------------
// WMMA GEMM: C[M,N] = A[M,K] @ Wt[N,K]^T + bias
//   mode 0: f16 row-major out      (Q/K projections)
//   mode 1: f32 row-major out      (final O projection -> d_out)
//   mode 2: f16 transposed-per-seq (V: out[(b*N+col)*SEQ + s], s = row % SEQ)
// 256 threads = 8 waves; block tile 128x128; wave tile 16x128; K step 32.
// B tile staged in double-buffered LDS (fragment-ready [col][k] layout,
// pitch 40 halves for bank-clean half-wave b128 reads), copied with the
// CDNA5 async global->LDS path when available (ASYNCcnt), else sync loads.
// -------------------------------------------------------------------------
#define BPITCH 40

__device__ __forceinline__ void stage_b(const _Float16* __restrict__ Wt,
                                        _Float16* __restrict__ ldsBuf,
                                        int colBase, int K, int kb, int tid) {
    const int c  = tid >> 1;              // 0..127
    const int ko = (tid & 1) * 16;        // 0 or 16
    const _Float16* src = Wt + (size_t)(colBase + c) * K + kb + ko;
    _Float16* dst = ldsBuf + c * BPITCH + ko;
#if ASYNC_LDS
    __builtin_amdgcn_global_load_async_to_lds_b128((gv4i*)src,       (lv4i*)dst,       0, 0);
    __builtin_amdgcn_global_load_async_to_lds_b128((gv4i*)(src + 8), (lv4i*)(dst + 8), 0, 0);
#else
    uint4 w0 = *(const uint4*)(src);
    uint4 w1 = *(const uint4*)(src + 8);
    *(uint4*)(dst)     = w0;
    *(uint4*)(dst + 8) = w1;
#endif
}

__global__ __launch_bounds__(256)
void gemm_wmma(const _Float16* __restrict__ A, const _Float16* __restrict__ Wt,
               const float* __restrict__ bias, void* __restrict__ Cout,
               int M, int N, int K, int mode) {
    __shared__ __align__(16) _Float16 ldsB[2][128][BPITCH];

    const int tid  = threadIdx.x;
    const int lane = tid & 31;
    const int wave = tid >> 5;
    const int lr   = lane & 15;
    const bool hiL = lane >= 16;
    const int sub  = hiL ? 8 : 0;     // A-fragment K-shift for upper half-wave

    const int rowBase = blockIdx.y * 128 + wave * 16;
    const int colBase = blockIdx.x * 128;

    v8f acc[8];
#pragma unroll
    for (int i = 0; i < 8; i++) acc[i] = {};

    const _Float16* aRow = A + (size_t)(rowBase + lr) * K;
    const int nsteps = K >> 5;

    // prologue: stage first B tile, preload first A fragment
    stage_b(Wt, &ldsB[0][0][0], colBase, K, 0, tid);
    Frag16 afCur;
    afCur.q[0] = *(const uint4*)(aRow + sub);
    afCur.q[1] = *(const uint4*)(aRow + sub + 16);

    for (int i = 0; i < nsteps; i++) {
#if ASYNC_LDS
        __builtin_amdgcn_s_wait_asynccnt(0);   // buf (i&1) copies landed in LDS
#endif
        __syncthreads();                       // visible to all waves; prior reads done
        if (i + 1 < nsteps)
            stage_b(Wt, &ldsB[(i + 1) & 1][0][0], colBase, K, (i + 1) << 5, tid);

        // prefetch next A fragment (global) while this step computes
        Frag16 afNext;
        if (i + 1 < nsteps) {
            const _Float16* ap = aRow + ((i + 1) << 5) + sub;
            afNext.q[0] = *(const uint4*)(ap);
            afNext.q[1] = *(const uint4*)(ap + 16);
        }

        // gather all 8 B fragments into distinct registers
        Frag16 bf[8];
#pragma unroll
        for (int ct = 0; ct < 8; ct++) {
            const _Float16* bp = &ldsB[i & 1][ct * 16 + lr][hiL ? 8 : 0];
            bf[ct].q[0] = *(const uint4*)(bp);
            bf[ct].q[1] = *(const uint4*)(bp + 16);
        }
        SCHED_BARRIER();   // keep the load batch ahead of the WMMA chain
#pragma unroll
        for (int ct = 0; ct < 8; ct++)
            acc[ct] = __builtin_amdgcn_wmma_f32_16x16x32_f16(
                false, afCur.v, false, bf[ct].v, (short)0, acc[ct], false, false);

        afCur = afNext;
    }

#pragma unroll
    for (int ct = 0; ct < 8; ct++) {
        const int col = colBase + ct * 16 + lr;
        const float bv = bias[col];
#pragma unroll
        for (int i = 0; i < 8; i++) {
            const int r = rowBase + (hiL ? i + 8 : i);
            const float val = acc[ct][i] + bv;
            if (mode == 0) {
                ((_Float16*)Cout)[(size_t)r * N + col] = (_Float16)val;
            } else if (mode == 1) {
                ((float*)Cout)[(size_t)r * N + col] = val;
            } else {
                const int bb = r >> 10, s = r & (SEQ - 1);
                ((_Float16*)Cout)[(((size_t)(bb * N + col)) << 10) + s] = (_Float16)val;
            }
        }
    }
}

// -------------------------------------------------------------------------
// Causal GQA flash attention (one wave per 16-row Q tile, full D=128).
// -------------------------------------------------------------------------
__global__ __launch_bounds__(128)
void attn_kernel(const _Float16* __restrict__ Q, const _Float16* __restrict__ Kc,
                 const _Float16* __restrict__ Vt, _Float16* __restrict__ Out) {
    __shared__ __align__(16) _Float16 ldsP[4][16][32];   // per-wave P staging

    const int lane = threadIdx.x & 31;
    const int wave = threadIdx.x >> 5;
    const int lr   = lane & 15;
    const bool hiL = lane >= 16;
    const int sub  = hiL ? 8 : 0;

    const int bid = blockIdx.x;
    const int grp = bid & 15;          // 16 groups of 4 q-tiles
    const int hq  = (bid >> 4) & 15;
    const int b   = bid >> 8;
    const int kvh = hq >> 2;           // G = 4
    const int qBase = (grp * 4 + wave) * 16;

    // Q fragments: 16 rows x 128 d, as 4 A-fragments of 16x32
    Frag16 qf[4];
    const _Float16* qp = Q + (size_t)(b * SEQ + qBase + lr) * (NQH * DH) + hq * DH;
#pragma unroll
    for (int c = 0; c < 4; c++) {
        qf[c].q[0] = *(const uint4*)(qp + c * 32 + sub);
        qf[c].q[1] = *(const uint4*)(qp + c * 32 + 16 + sub);
    }

    v8f o[8];
    float m[8], l[8];
#pragma unroll
    for (int i = 0; i < 8; i++) { o[i] = {}; m[i] = -3.0e38f; l[i] = 0.0f; }

    const float scale = 0.08838834764831845f;   // 1/sqrt(128)
    const int nc = (qBase + 47) >> 5;           // 32-key chunks needed (causal)

    const _Float16* kBaseP = Kc + (size_t)(b * SEQ + lr) * (NKVH * DH)
                             + kvh * DH + (hiL ? 8 : 0);
    const _Float16* vBaseP = Vt + (size_t)((b * NKVH + kvh) * DH + lr) * SEQ
                             + (hiL ? 8 : 0);

    for (int kc = 0; kc < nc; kc++) {
        const int kb0 = kc * 32;

        // ---- batch-load all 8 K^T fragments (2 key tiles x 4 d-chunks) ----
        Frag16 kf[2][4];
#pragma unroll
        for (int t = 0; t < 2; t++) {
            const _Float16* kp = kBaseP + (size_t)(kb0 + t * 16) * (NKVH * DH);
#pragma unroll
            for (int c = 0; c < 4; c++) {
                kf[t][c].q[0] = *(const uint4*)(kp + c * 32);
                kf[t][c].q[1] = *(const uint4*)(kp + c * 32 + 16);
            }
        }
        SCHED_BARRIER();

        // ---- scores: two 16x16 tiles, K-dim = D = 128 (4 WMMAs each) ----
        v8f s[2];
#pragma unroll
        for (int t = 0; t < 2; t++) {
            v8f st = {};
#pragma unroll
            for (int c = 0; c < 4; c++)
                st = __builtin_amdgcn_wmma_f32_16x16x32_f16(
                    false, qf[c].v, false, kf[t][c].v, (short)0, st, false, false);
            s[t] = st;
        }

        // ---- batch-load all 8 V fragments (overlap with softmax VALU) ----
        Frag16 vf[8];
#pragma unroll
        for (int ct = 0; ct < 8; ct++) {
            const _Float16* vp = vBaseP + (size_t)(ct * 16) * SEQ + kb0;
            vf[ct].q[0] = *(const uint4*)(vp);
            vf[ct].q[1] = *(const uint4*)(vp + 16);
        }
        SCHED_BARRIER();

        // ---- scale + causal mask ----
#pragma unroll
        for (int t = 0; t < 2; t++) {
            const int key = kb0 + t * 16 + lr;
#pragma unroll
            for (int i = 0; i < 8; i++) {
                const int qr = qBase + (hiL ? i + 8 : i);
                float v = s[t][i] * scale;
                s[t][i] = (key > qr) ? -3.0e38f : v;
            }
        }

        // ---- online softmax update ----
#pragma unroll
        for (int i = 0; i < 8; i++) {
            float mt = fmaxf(s[0][i], s[1][i]);
#pragma unroll
            for (int off = 1; off < 16; off <<= 1)
                mt = fmaxf(mt, __shfl_xor(mt, off, 32));
            const float mn    = fmaxf(m[i], mt);
            const float alpha = __expf(m[i] - mn);
            const float p0 = __expf(s[0][i] - mn);
            const float p1 = __expf(s[1][i] - mn);
            float rs = p0 + p1;
#pragma unroll
            for (int off = 1; off < 16; off <<= 1)
                rs += __shfl_xor(rs, off, 32);
            l[i] = l[i] * alpha + rs;
            m[i] = mn;
#pragma unroll
            for (int ct = 0; ct < 8; ct++) o[ct][i] = o[ct][i] * alpha;

            const int row = hiL ? i + 8 : i;
            ldsP[wave][row][lr]      = (_Float16)p0;   // cols kb0+lr
            ldsP[wave][row][16 + lr] = (_Float16)p1;   // cols kb0+16+lr
        }

        // ---- re-layout P as a 16x32 A-fragment via per-wave LDS (in-order) ----
        Frag16 pf;
        const _Float16* pp = &ldsP[wave][lr][0];
        pf.q[0] = *(const uint4*)(pp + sub);
        pf.q[1] = *(const uint4*)(pp + 16 + sub);

        // ---- PV: O(16x128) += P(16x32) @ V(32x128), 8 WMMAs ----
#pragma unroll
        for (int ct = 0; ct < 8; ct++)
            o[ct] = __builtin_amdgcn_wmma_f32_16x16x32_f16(
                false, pf.v, false, vf[ct].v, (short)0, o[ct], false, false);
    }

    // ---- normalize and store f16 attention output (token-major) ----
    float inv[8];
#pragma unroll
    for (int i = 0; i < 8; i++) inv[i] = 1.0f / l[i];
#pragma unroll
    for (int ct = 0; ct < 8; ct++) {
#pragma unroll
        for (int i = 0; i < 8; i++) {
            const int r = qBase + (hiL ? i + 8 : i);
            Out[(size_t)(b * SEQ + r) * (NQH * DH) + hq * DH + ct * 16 + lr] =
                (_Float16)(o[ct][i] * inv[i]);
        }
    }
}

// -------------------------------------------------------------------------
// Host-side launch
// -------------------------------------------------------------------------
extern "C" void kernel_launch(void* const* d_in, const int* in_sizes, int n_in,
                              void* d_out, int out_size, void* d_ws, size_t ws_size,
                              hipStream_t stream) {
    (void)in_sizes; (void)n_in; (void)out_size; (void)ws_size;
    const float* x  = (const float*)d_in[0];
    const float* Wq = (const float*)d_in[1];
    const float* bq = (const float*)d_in[2];
    const float* Wk = (const float*)d_in[3];
    const float* bk = (const float*)d_in[4];
    const float* Wv = (const float*)d_in[5];
    const float* bv = (const float*)d_in[6];
    const float* Wo = (const float*)d_in[7];
    const float* bo = (const float*)d_in[8];
    float* out = (float*)d_out;

    char* ws = (char*)d_ws;
    _Float16* xh  = (_Float16*)(ws);                    // 32 MB  [NTOK][HID]
    _Float16* qh  = (_Float16*)(ws + 33554432);         // 32 MB  [NTOK][NQH*DH]
    _Float16* kh  = (_Float16*)(ws + 67108864);         // 8 MB   [NTOK][NKVH*DH]
    _Float16* vT  = (_Float16*)(ws + 75497472);         // 8 MB   transposed V
    _Float16* at  = (_Float16*)(ws + 83886080);         // 32 MB  attention out
    _Float16* WqT = (_Float16*)(ws + 117440512);        // 8 MB   [2048][2048]
    _Float16* WkT = (_Float16*)(ws + 125829120);        // 2 MB   [512][2048]
    _Float16* WvT = (_Float16*)(ws + 127926272);        // 2 MB   [512][2048]
    _Float16* WoT = (_Float16*)(ws + 130023424);        // 8 MB   [2048][2048]

    cvt_f16k<<<4096, 256, 0, stream>>>(x, xh, NTOK * HID);

    dim3 tb(32, 8);
    transpose_cvt<<<dim3(64, 64), tb, 0, stream>>>(Wq, WqT, HID, NQH * DH);
    transpose_cvt<<<dim3(16, 64), tb, 0, stream>>>(Wk, WkT, HID, NKVH * DH);
    transpose_cvt<<<dim3(16, 64), tb, 0, stream>>>(Wv, WvT, HID, NKVH * DH);
    transpose_cvt<<<dim3(64, 64), tb, 0, stream>>>(Wo, WoT, NQH * DH, HID);

    // Q/K/V projections
    gemm_wmma<<<dim3(16, 64), 256, 0, stream>>>(xh, WqT, bq, qh, NTOK, NQH * DH, HID, 0);
    gemm_wmma<<<dim3(4, 64), 256, 0, stream>>>(xh, WkT, bk, kh, NTOK, NKVH * DH, HID, 0);
    gemm_wmma<<<dim3(4, 64), 256, 0, stream>>>(xh, WvT, bv, vT, NTOK, NKVH * DH, HID, 2);

    // causal GQA attention
    attn_kernel<<<NB * NQH * 16, 128, 0, stream>>>(qh, kh, vT, at);

    // output projection -> f32 d_out
    gemm_wmma<<<dim3(16, 64), 256, 0, stream>>>(at, WoT, bo, out, NTOK, HID, NQH * DH, 1);
}
